// LaCTSWIGLULayer_23012434772142
// MI455X (gfx1250) — compile-verified, matrix operand order checked by reading, not run on hardware
//
#include <hip/hip_runtime.h>
#include <math.h>

// ---------------------------------------------------------------------------
// Problem constants (from the reference)
// ---------------------------------------------------------------------------
#define SEQ     8192
#define HID     2048
#define NHEAD   16
#define HDIM    128
#define NFW     4
#define FDIM    512
#define CHUNK   2048
#define WIN     256

typedef unsigned short ush;
typedef __attribute__((ext_vector_type(16))) __bf16        v16bf;
typedef __attribute__((ext_vector_type(8)))  float         v8f;
typedef __attribute__((ext_vector_type(4)))  unsigned int  u32x4;

union Frag16 {
    v16bf v;
    ush   u[16];
    u32x4 q[2];
};

__device__ inline ush f2bf(float x) {
    unsigned u = __float_as_uint(x);
    unsigned r = (u + 0x7FFFu + ((u >> 16) & 1u)) >> 16;
    return (ush)r;
}
__device__ inline float bf2f(ush h) {
    return __uint_as_float(((unsigned)h) << 16);
}

// ---------------------------------------------------------------------------
// WMMA 16x16x32 bf16 fragment helpers.
// ISA layout (16-bit A-matrix 16x32): lane = row (lane&15),
//   lanes 0-15 : elements 0..7 -> K = 0..7,  8..15 -> K = 16..23
//   lanes 16-31: elements 0..7 -> K = 8..15, 8..15 -> K = 24..31
// B loaded symmetrically from an [N,K] row-major matrix (NT GEMM).
// frag_ptr: per-lane row pointer (row + K-phase baked in).
// frag_at : load at a constant element offset -> immediate-offset b128 loads.
// ---------------------------------------------------------------------------
__device__ inline const ush* frag_ptr(const ush* __restrict__ base,
                                      long rs, int r, int lane)
{
    return base + (long)(r + (lane & 15)) * rs + ((lane >> 4) << 3);
}

__device__ inline v16bf frag_at(const ush* __restrict__ p, int koff)
{
    Frag16 f;
    f.q[0] = *(const u32x4*)(p + koff);
    f.q[1] = *(const u32x4*)(p + koff + 16);
    return f.v;
}

// legacy-style loader (used by the attention kernel, addresses vary per call)
__device__ inline v16bf frag_load(const ush* __restrict__ base,
                                  long rs, int r, int k0, int lane)
{
    return frag_at(frag_ptr(base, rs, r, lane), k0);
}

// ---------------------------------------------------------------------------
// Batched NT GEMM:  C[m,n] (+)= alpha * sum_k A[m*lda+k] * B[n*ldb+k]  (bf16)
// C row-major fp32.  M mult of 128, N mult of 64, K mult of 64.
// Block = 128 threads (4 waves); block tile 128x64; wave tile 32x64
// (2 A-frags x 4 B-frags = 8 WMMA per K-step, B reused across A rows).
// Software-pipelined with two fragment banks; all fragment loads are
// immediate-offset b128 loads off six loop-carried row pointers, so there is
// no per-iteration address recomputation (and no spills).
// ---------------------------------------------------------------------------
__global__ void __launch_bounds__(128)
k_gemm(const ush* __restrict__ A, long lda, long bsA,
       const ush* __restrict__ B, long ldb, long bsB,
       float* __restrict__ C, long ldc, long bsC,
       int M, int N, int K, float alpha, int accum)
{
    int b = blockIdx.z;
    A += (long)b * bsA;  B += (long)b * bsB;  C += (long)b * bsC;
    int bm   = blockIdx.x * 128;
    int bn   = blockIdx.y * 64;
    int wave = threadIdx.x >> 5;
    int lane = threadIdx.x & 31;
    int m0   = bm + wave * 32;

    // six loop-carried row pointers (K-phase per lane already included)
    const ush* pa0 = frag_ptr(A, lda, m0,      lane);
    const ush* pa1 = frag_ptr(A, lda, m0 + 16, lane);
    const ush* pb0 = frag_ptr(B, ldb, bn,      lane);
    const ush* pb1 = frag_ptr(B, ldb, bn + 16, lane);
    const ush* pb2 = frag_ptr(B, ldb, bn + 32, lane);
    const ush* pb3 = frag_ptr(B, ldb, bn + 48, lane);

    v8f acc[2][4] = {};
    v16bf a0c, a1c, b0c[4];     // bank 0
    v16bf a0n, a1n, b0n[4];     // bank 1

    // prologue: bank 0 <- first 32 K
    a0c    = frag_at(pa0, 0);
    a1c    = frag_at(pa1, 0);
    b0c[0] = frag_at(pb0, 0);
    b0c[1] = frag_at(pb1, 0);
    b0c[2] = frag_at(pb2, 0);
    b0c[3] = frag_at(pb3, 0);

    for (int k0 = 0; k0 < K - 64; k0 += 64) {
        // bank 1 <- next 32 K (in flight while bank-0 WMMAs run)
        a0n    = frag_at(pa0, 32);
        a1n    = frag_at(pa1, 32);
        b0n[0] = frag_at(pb0, 32);
        b0n[1] = frag_at(pb1, 32);
        b0n[2] = frag_at(pb2, 32);
        b0n[3] = frag_at(pb3, 32);
        __builtin_prefetch(pa0 + 64, 0, 1);
#pragma unroll
        for (int t = 0; t < 4; ++t) {
            acc[0][t] = __builtin_amdgcn_wmma_f32_16x16x32_bf16(
                false, a0c, false, b0c[t], (short)0, acc[0][t], false, false);
            acc[1][t] = __builtin_amdgcn_wmma_f32_16x16x32_bf16(
                false, a1c, false, b0c[t], (short)0, acc[1][t], false, false);
        }
        pa0 += 64; pa1 += 64;
        pb0 += 64; pb1 += 64; pb2 += 64; pb3 += 64;
        // bank 0 <- following 32 K (in flight while bank-1 WMMAs run)
        a0c    = frag_at(pa0, 0);
        a1c    = frag_at(pa1, 0);
        b0c[0] = frag_at(pb0, 0);
        b0c[1] = frag_at(pb1, 0);
        b0c[2] = frag_at(pb2, 0);
        b0c[3] = frag_at(pb3, 0);
#pragma unroll
        for (int t = 0; t < 4; ++t) {
            acc[0][t] = __builtin_amdgcn_wmma_f32_16x16x32_bf16(
                false, a0n, false, b0n[t], (short)0, acc[0][t], false, false);
            acc[1][t] = __builtin_amdgcn_wmma_f32_16x16x32_bf16(
                false, a1n, false, b0n[t], (short)0, acc[1][t], false, false);
        }
    }
    // epilogue: last two K-steps
    a0n    = frag_at(pa0, 32);
    a1n    = frag_at(pa1, 32);
    b0n[0] = frag_at(pb0, 32);
    b0n[1] = frag_at(pb1, 32);
    b0n[2] = frag_at(pb2, 32);
    b0n[3] = frag_at(pb3, 32);
#pragma unroll
    for (int t = 0; t < 4; ++t) {
        acc[0][t] = __builtin_amdgcn_wmma_f32_16x16x32_bf16(
            false, a0c, false, b0c[t], (short)0, acc[0][t], false, false);
        acc[1][t] = __builtin_amdgcn_wmma_f32_16x16x32_bf16(
            false, a1c, false, b0c[t], (short)0, acc[1][t], false, false);
    }
#pragma unroll
    for (int t = 0; t < 4; ++t) {
        acc[0][t] = __builtin_amdgcn_wmma_f32_16x16x32_bf16(
            false, a0n, false, b0n[t], (short)0, acc[0][t], false, false);
        acc[1][t] = __builtin_amdgcn_wmma_f32_16x16x32_bf16(
            false, a1n, false, b0n[t], (short)0, acc[1][t], false, false);
    }

    int col = lane & 15;
    int mo  = (lane >> 4) << 3;     // C layout: lane<16 -> rows 0..7, else 8..15
#pragma unroll
    for (int r = 0; r < 2; ++r) {
#pragma unroll
        for (int t = 0; t < 4; ++t) {
#pragma unroll
            for (int i = 0; i < 8; ++i) {
                long idx = (long)(m0 + r * 16 + mo + i) * ldc + bn + t * 16 + col;
                float v = alpha * acc[r][t][i];
                if (accum) C[idx] += v; else C[idx] = v;
            }
        }
    }
}

// ---------------------------------------------------------------------------
// fp32 -> bf16 cast (grid-strided), and per-batch transposed cast
// ---------------------------------------------------------------------------
__global__ void k_cast_bf16(const float* __restrict__ src, ush* __restrict__ dst, long n)
{
    long i = (long)blockIdx.x * blockDim.x + threadIdx.x;
    long st = (long)gridDim.x * blockDim.x;
    for (; i < n; i += st) dst[i] = f2bf(src[i]);
}

// dst[b][j][i] = bf16(src[b][i][j])   for [NFW][FDIM][FDIM]
__global__ void k_cast_t(const float* __restrict__ src, ush* __restrict__ dst)
{
    long idx = (long)blockIdx.x * blockDim.x + threadIdx.x;
    long n = (long)NFW * FDIM * FDIM;
    if (idx >= n) return;
    int j = (int)(idx & (FDIM - 1));
    long bi = idx >> 9;
    int i = (int)(bi & (FDIM - 1));
    int b = (int)(bi >> 9);
    dst[((long)b * FDIM + j) * FDIM + i] = f2bf(src[idx]);
}

// ---------------------------------------------------------------------------
// Per-token learning rates: lr[s,j] = softplus(hs[s,:]·lr_w[j,:] + lr_b[j] + base)
// ---------------------------------------------------------------------------
__global__ void k_lr(const float* __restrict__ hs, const float* __restrict__ lrw,
                     const float* __restrict__ lrb, float* __restrict__ out, float base)
{
    int s = blockIdx.x, t = threadIdx.x;
    float p[12];
#pragma unroll
    for (int j = 0; j < 12; ++j) p[j] = 0.f;
    const float* x = hs + (long)s * HID;
    for (int d = t; d < HID; d += 128) {
        float xv = x[d];
#pragma unroll
        for (int j = 0; j < 12; ++j) p[j] += xv * lrw[(long)j * HID + d];
    }
    __shared__ float red[12][128];
#pragma unroll
    for (int j = 0; j < 12; ++j) red[j][t] = p[j];
    __syncthreads();
    for (int st = 64; st > 0; st >>= 1) {
        if (t < st)
#pragma unroll
            for (int j = 0; j < 12; ++j) red[j][t] += red[j][t + st];
        __syncthreads();
    }
    if (t < 12) {
        float v = red[t][0] + lrb[t] + base;
        out[(long)s * 12 + t] = (v > 20.f) ? v : log1pf(__expf(v));
    }
}

// ---------------------------------------------------------------------------
// Per-token postprocess: RMS-norm q/k, RoPE (bf16), transposed V (bf16),
// SiLU + per-head l2 norm for the fast-weight path; fk also stored transposed.
// One block (256 thr) per token.
// ---------------------------------------------------------------------------
__global__ void k_post(const float* __restrict__ qkv,
                       const float* __restrict__ qnw, const float* __restrict__ knw,
                       const float* __restrict__ qks, const float* __restrict__ qko,
                       ush* __restrict__ qrot, ush* __restrict__ krot,
                       ush* __restrict__ vt,
                       ush* __restrict__ fq, ush* __restrict__ fk,
                       ush* __restrict__ fkT, ush* __restrict__ fv)
{
    int s = blockIdx.x, t = threadIdx.x;
    const float* base = qkv + (long)s * (3 * HID);
    __shared__ float sh[HID];
    __shared__ float sh2[HID];
    __shared__ float red[256];
    __shared__ float hn[NFW];

    for (int pass = 0; pass < 2; ++pass) {
        const float* src = base + pass * HID;
        const float* nw  = pass ? knw : qnw;
        float p = 0.f;
        for (int i = t; i < HID; i += 256) { float x = src[i]; p += x * x; }
        red[t] = p; __syncthreads();
        for (int st = 128; st > 0; st >>= 1) { if (t < st) red[t] += red[t + st]; __syncthreads(); }
        float rinv = rsqrtf(red[0] / (float)HID + 1e-6f);
        __syncthreads();
        for (int i = t; i < HID; i += 256) sh[i] = src[i] * rinv * nw[i];
        __syncthreads();
        // RoPE (theta = 5e5, head dim 128, half-rotation form)
        ush* rout = pass ? krot : qrot;
        for (int i = t; i < HID; i += 256) {
            int d = i & (HDIM - 1);
            int j = d & 63;
            float f = (float)s * __powf(500000.0f, -((float)(2 * j)) / 128.0f);
            float sn, cs_; __sincosf(f, &sn, &cs_);
            float val = (d < 64) ? (sh[i] * cs_ - sh[i + 64] * sn)
                                 : (sh[i] * cs_ + sh[i - 64] * sn);
            rout[(long)s * HID + i] = f2bf(val);
        }
        // fast-weight path: silu(q*scale+off), per-head l2 normalize
        for (int i = t; i < HID; i += 256) {
            float x  = sh[i] * qks[2 * i + pass] + qko[2 * i + pass];
            float sg = 1.f / (1.f + __expf(-x));
            sh2[i] = x * sg;
        }
        __syncthreads();
        int hh = t >> 6, l = t & 63;
        float pp = 0.f;
        for (int j = l; j < FDIM; j += 64) { float x = sh2[hh * FDIM + j]; pp += x * x; }
        red[t] = pp; __syncthreads();
        for (int st = 32; st > 0; st >>= 1) { if (l < st) red[t] += red[t + st]; __syncthreads(); }
        if (l == 0) hn[hh] = fmaxf(sqrtf(red[t]), 1e-12f);
        __syncthreads();
        ush* fo = pass ? fk : fq;
        for (int i = t; i < HID; i += 256) {
            int head = i >> 9, d = i & (FDIM - 1);
            ush bv = f2bf(sh2[i] / hn[head]);
            fo[((long)head * SEQ + s) * FDIM + d] = bv;
            if (pass) fkT[((long)head * FDIM + d) * SEQ + s] = bv;
        }
        __syncthreads();
    }
    // V: raw (transposed for attention) and silu (fast-weight path)
    const float* vsrc = base + 2 * HID;
    for (int i = t; i < HID; i += 256) {
        float x = vsrc[i];
        vt[(long)i * SEQ + s] = f2bf(x);
        float sg = 1.f / (1.f + __expf(-x));
        int head = i >> 9, d = i & (FDIM - 1);
        fv[((long)head * SEQ + s) * FDIM + d] = f2bf(x * sg);
    }
}

// ---------------------------------------------------------------------------
// Sliding-window attention (window 256, causal). One wave per (head, 16-query
// tile). QK^T + masked softmax + PV, all matrix ops through WMMA.
// ---------------------------------------------------------------------------
__global__ void k_attn(const ush* __restrict__ qrot, const ush* __restrict__ krot,
                       const ush* __restrict__ vt, float* __restrict__ attn_o)
{
    __shared__ float sc[16][320];
    __shared__ ush   pb[16][320];
    int qt = blockIdx.x, h = blockIdx.y;
    int lane = threadIdx.x;
    int qs = qt << 4;
    int kw0 = qs - (WIN - 1); if (kw0 < 0) kw0 = 0; kw0 &= ~31;
    int cols   = qs + 16 - kw0;          // <= 302, multiple of 16
    int nkt    = cols >> 4;
    int ksteps = (cols + 31) >> 5;

    for (int i = lane; i < 16 * 320; i += 32) { (&sc[0][0])[i] = 0.f; (&pb[0][0])[i] = 0; }
    __syncthreads();

    v16bf qf[4];
#pragma unroll
    for (int kk = 0; kk < 4; ++kk)
        qf[kk] = frag_load(qrot + h * HDIM, HID, qs, kk * 32, lane);

    const float scale = 0.088388347648318447f;  // 1/sqrt(128)
    for (int kt = 0; kt < nkt; ++kt) {
        v8f acc = {};
#pragma unroll
        for (int kk = 0; kk < 4; ++kk) {
            v16bf kf = frag_load(krot + h * HDIM, HID, kw0 + kt * 16, kk * 32, lane);
            acc = __builtin_amdgcn_wmma_f32_16x16x32_bf16(
                false, qf[kk], false, kf, (short)0, acc, false, false);
        }
        int col = lane & 15, mo = (lane >> 4) << 3;
#pragma unroll
        for (int i = 0; i < 8; ++i) {
            int m = mo + i, qpos = qs + m, kpos = kw0 + kt * 16 + col;
            float v = acc[i] * scale;
            if (kpos > qpos || kpos < qpos - (WIN - 1)) v = -1e30f;
            sc[m][kt * 16 + col] = v;
        }
    }
    __syncthreads();
    if (lane < 16) {
        int m = lane;
        float mx = -1e30f;
        for (int j = 0; j < cols; ++j) mx = fmaxf(mx, sc[m][j]);
        float sum = 0.f;
        for (int j = 0; j < cols; ++j) { float e = __expf(sc[m][j] - mx); sc[m][j] = e; sum += e; }
        float inv = 1.f / sum;
        for (int j = 0; j < cols; ++j) pb[m][j] = f2bf(sc[m][j] * inv);
    }
    __syncthreads();
    // O = P @ V  (V transposed buffer: rows = h*128+d, cols = seq)
    const ush* vb = vt + (long)h * HDIM * SEQ + kw0;
#pragma unroll
    for (int t2 = 0; t2 < 8; ++t2) {
        v8f acc = {};
        for (int st = 0; st < ksteps; ++st) {
            v16bf pf = frag_load(&pb[0][0], 320, 0, st * 32, lane);
            v16bf vf = frag_load(vb, SEQ, t2 * 16, st * 32, lane);
            acc = __builtin_amdgcn_wmma_f32_16x16x32_bf16(
                false, pf, false, vf, (short)0, acc, false, false);
        }
        int col = lane & 15, mo = (lane >> 4) << 3;
#pragma unroll
        for (int i = 0; i < 8; ++i)
            attn_o[(long)(qs + mo + i) * HID + h * HDIM + t2 * 16 + col] = acc[i];
    }
}

// ---------------------------------------------------------------------------
// TTT elementwise pieces
// ---------------------------------------------------------------------------
__global__ void k_silu_mul(const float* __restrict__ a, const float* __restrict__ b,
                           ush* __restrict__ out, long n)
{
    long i = (long)blockIdx.x * blockDim.x + threadIdx.x;
    if (i >= n) return;
    float g = a[i];
    float sg = 1.f / (1.f + __expf(-g));
    out[i] = f2bf(g * sg * b[i]);
}

// From g, hm, dhid compute hid and the lr-scaled update operands; all four
// outputs are stored TRANSPOSED [NFW][FDIM][CHUNK] so the weight-update GEMMs
// read contiguous K fragments (no gathers anywhere).
__global__ void k_ttt_grads(const float* __restrict__ g, const float* __restrict__ hm,
                            const float* __restrict__ dhid, const ush* __restrict__ fv,
                            const float* __restrict__ lr, int chunk,
                            ush* __restrict__ hidT, ush* __restrict__ a0T,
                            ush* __restrict__ a1T, ush* __restrict__ a2T)
{
    long idx = (long)blockIdx.x * blockDim.x + threadIdx.x;
    long n = (long)NFW * CHUNK * FDIM;
    if (idx >= n) return;
    int d   = (int)(idx & (FDIM - 1));
    long cd = idx >> 9;
    int tok = (int)(cd & (CHUNK - 1));
    int hh  = (int)(cd >> 11);
    int sg_ = chunk * CHUNK + tok;
    float l0 = lr[(long)sg_ * 12 + hh];
    float l1 = lr[(long)sg_ * 12 + 4 + hh];
    float l2 = lr[(long)sg_ * 12 + 8 + hh];
    float gg = g[idx], hmv = hm[idx], dh = dhid[idx];
    float sg = 1.f / (1.f + __expf(-gg));
    float silu_g = gg * sg;
    float hid = silu_g * hmv;
    float dhm = dh * silu_g;
    float dg  = dh * hmv;
    float dgp = dg * (sg * (1.f + gg * (1.f - sg)));
    long ti = ((long)hh * FDIM + d) * CHUNK + tok;    // transposed index
    hidT[ti] = f2bf(hid);
    a0T[ti]  = f2bf(dgp * l0);
    a2T[ti]  = f2bf(dhm * l2);
    float vv = bf2f(fv[((long)hh * SEQ + sg_) * FDIM + d]);
    a1T[ti]  = f2bf(vv * l1);
}

// ---------------------------------------------------------------------------
// TTT output RMS-norm + combine with attention, cast to bf16 for final GEMM.
// ---------------------------------------------------------------------------
__global__ void k_comb(const float* __restrict__ o_fast, const float* __restrict__ attn_o,
                       const float* __restrict__ tnw, ush* __restrict__ xo)
{
    int s = blockIdx.x, hh = blockIdx.y, t = threadIdx.x;
    const float* o = o_fast + ((long)hh * SEQ + s) * FDIM;
    float p = 0.f;
    for (int i = t; i < FDIM; i += 128) { float x = o[i]; p += x * x; }
    __shared__ float red[128];
    red[t] = p; __syncthreads();
    for (int st = 64; st > 0; st >>= 1) { if (t < st) red[t] += red[t + st]; __syncthreads(); }
    float rinv = rsqrtf(red[0] / (float)FDIM + 1e-6f);
    for (int i = t; i < FDIM; i += 128) {
        float y = o[i] * rinv * tnw[i] + attn_o[(long)s * HID + hh * FDIM + i];
        xo[(long)s * HID + hh * FDIM + i] = f2bf(y);
    }
}

// ---------------------------------------------------------------------------
// Host orchestration
// ---------------------------------------------------------------------------
static inline void gemm(hipStream_t st,
                        const ush* A, long lda, long bsA,
                        const ush* B, long ldb, long bsB,
                        float* C, long ldc, long bsC,
                        int M, int N, int K, int batch, float alpha, int accum)
{
    dim3 g(M / 128, N / 64, batch);
    k_gemm<<<g, 128, 0, st>>>(A, lda, bsA, B, ldb, bsB, C, ldc, bsC,
                              M, N, K, alpha, accum);
}

extern "C" void kernel_launch(void* const* d_in, const int* in_sizes, int n_in,
                              void* d_out, int out_size, void* d_ws, size_t ws_size,
                              hipStream_t stream)
{
    const float* hs   = (const float*)d_in[0];
    const float* qkvw = (const float*)d_in[1];
    const float* qnw  = (const float*)d_in[2];
    const float* knw  = (const float*)d_in[3];
    const float* qks  = (const float*)d_in[4];
    const float* qko  = (const float*)d_in[5];
    const float* lrw  = (const float*)d_in[6];
    const float* lrb  = (const float*)d_in[7];
    const float* w0in = (const float*)d_in[8];
    const float* w1in = (const float*)d_in[9];
    const float* w2in = (const float*)d_in[10];
    const float* tnw  = (const float*)d_in[11];
    const float* opw  = (const float*)d_in[12];
    float* out = (float*)d_out;

    char* ws = (char*)d_ws;
    size_t off = 0;
    auto alloc = [&](size_t bytes) -> void* {
        void* p = (void*)(ws + off);
        off += bytes; off = (off + 255) & ~(size_t)255;
        return p;
    };
    const long SH  = (long)SEQ * HID;           // 16.7M
    const long FWN = (long)NFW * SEQ * FDIM;    // 16.7M
    const long WN  = (long)NFW * FDIM * FDIM;   // 1.05M
    const long CN  = (long)NFW * CHUNK * FDIM;  // 4.19M

    ush*   hs_bf   = (ush*)alloc(SH * 2);
    ush*   qkvw_bf = (ush*)alloc((long)3 * HID * HID * 2);
    ush*   ow_bf   = (ush*)alloc((long)HID * HID * 2);
    float* qkv_f   = (float*)alloc((long)SEQ * 3 * HID * 4);
    float* lr_f    = (float*)alloc((long)SEQ * 12 * 4);
    ush*   qrot    = (ush*)alloc(SH * 2);
    ush*   krot    = (ush*)alloc(SH * 2);
    ush*   vt      = (ush*)alloc(SH * 2);
    float* attn_o  = (float*)alloc(SH * 4);
    ush*   fq      = (ush*)alloc(FWN * 2);
    ush*   fk      = (ush*)alloc(FWN * 2);
    ush*   fkT     = (ush*)alloc(FWN * 2);        // [4][512][8192]
    ush*   fv      = (ush*)alloc(FWN * 2);
    float* wf      = (float*)alloc(3 * WN * 4);   // w0|w1|w2 fp32 masters
    ush*   wb      = (ush*)alloc(3 * WN * 2);     // bf16 snapshots per chunk
    ush*   w1bT    = (ush*)alloc(WN * 2);         // w1^T bf16 per chunk
    float* bufA    = (float*)alloc(CN * 4);
    float* bufB    = (float*)alloc(CN * 4);
    float* bufC    = (float*)alloc(CN * 4);
    ush*   hidq    = (ush*)alloc(CN * 2);
    ush*   hidT    = (ush*)alloc(CN * 2);         // [4][512][2048]
    ush*   a0T     = (ush*)alloc(CN * 2);
    ush*   a1T     = (ush*)alloc(CN * 2);
    ush*   a2T     = (ush*)alloc(CN * 2);
    float* o_fast  = (float*)alloc(FWN * 4);
    ush*   xo      = (ush*)alloc(SH * 2);
    (void)ws_size; (void)in_sizes; (void)n_in; (void)out_size;

    float* w0f = wf;   float* w1f = wf + WN;   float* w2f = wf + 2 * WN;
    ush*   w0b = wb;   ush*   w1b = wb + WN;   ush*   w2b = wb + 2 * WN;

    // ---- casts of static operands to bf16 --------------------------------
    k_cast_bf16<<<4096, 256, 0, stream>>>(hs, hs_bf, SH);
    k_cast_bf16<<<4096, 256, 0, stream>>>(qkvw, qkvw_bf, (long)3 * HID * HID);
    k_cast_bf16<<<2048, 256, 0, stream>>>(opw, ow_bf, (long)HID * HID);

    // ---- learning rates (tiny N=12 GEMM, VALU in fp32) -------------------
    const float base_lr_inv = logf(expm1f(0.001f));
    k_lr<<<SEQ, 128, 0, stream>>>(hs, lrw, lrb, lr_f, base_lr_inv);

    // ---- QKV projection: [8192,6144] = hs @ qkv_w^T  (WMMA) --------------
    gemm(stream, hs_bf, HID, 0, qkvw_bf, HID, 0,
         qkv_f, 3 * HID, 0, SEQ, 3 * HID, HID, 1, 1.0f, 0);

    // ---- per-token norms / RoPE / silu / l2 ------------------------------
    k_post<<<SEQ, 256, 0, stream>>>(qkv_f, qnw, knw, qks, qko,
                                    qrot, krot, vt, fq, fk, fkT, fv);

    // ---- sliding-window attention (WMMA) ---------------------------------
    k_attn<<<dim3(SEQ / 16, NHEAD), 32, 0, stream>>>(qrot, krot, vt, attn_o);

    // ---- TTT fast-weight recurrence --------------------------------------
    hipMemcpyAsync(w0f, w0in, WN * 4, hipMemcpyDeviceToDevice, stream);
    hipMemcpyAsync(w1f, w1in, WN * 4, hipMemcpyDeviceToDevice, stream);
    hipMemcpyAsync(w2f, w2in, WN * 4, hipMemcpyDeviceToDevice, stream);

    const long bsS  = (long)SEQ * FDIM;      // per-head stride, [4][8192][512]
    const long bsTS = (long)FDIM * SEQ;      // per-head stride, [4][512][8192]
    const long bsC_ = (long)CHUNK * FDIM;    // per-head stride, chunk buffers
    const long bsW  = (long)FDIM * FDIM;
    const int  NC   = (int)((CN + 255) / 256);
    for (int c = 0; c < 4; ++c) {
        long co = (long)c * CHUNK * FDIM;
        const ush* fqc  = fq  + co;
        const ush* fkc  = fk  + co;
        const ush* fvc  = fv  + co;
        const ush* fkTc = fkT + (long)c * CHUNK;   // column offset into [512][8192]
        // bf16 snapshots of current fast weights (+ w1 transposed)
        k_cast_bf16<<<2048, 256, 0, stream>>>(wf, wb, 3 * WN);
        k_cast_t<<<(int)((WN + 255) / 256), 256, 0, stream>>>(w1f, w1bT);
        // output path: o = (silu(q@w0^T) * (q@w2^T)) @ w1^T
        gemm(stream, fqc, FDIM, bsS, w0b, FDIM, bsW,
             bufA, FDIM, bsC_, CHUNK, FDIM, FDIM, NFW, 1.0f, 0);
        gemm(stream, fqc, FDIM, bsS, w2b, FDIM, bsW,
             bufB, FDIM, bsC_, CHUNK, FDIM, FDIM, NFW, 1.0f, 0);
        k_silu_mul<<<NC, 256, 0, stream>>>(bufA, bufB, hidq, CN);
        gemm(stream, hidq, FDIM, bsC_, w1b, FDIM, bsW,
             o_fast + co, FDIM, bsS, CHUNK, FDIM, FDIM, NFW, 1.0f, 0);
        // gradient path: g = k@w0^T, hm = k@w2^T, dhid = v@w1 (via w1^T)
        gemm(stream, fkc, FDIM, bsS, w0b, FDIM, bsW,
             bufA, FDIM, bsC_, CHUNK, FDIM, FDIM, NFW, 1.0f, 0);
        gemm(stream, fkc, FDIM, bsS, w2b, FDIM, bsW,
             bufB, FDIM, bsC_, CHUNK, FDIM, FDIM, NFW, 1.0f, 0);
        gemm(stream, fvc, FDIM, bsS, w1bT, FDIM, bsW,
             bufC, FDIM, bsC_, CHUNK, FDIM, FDIM, NFW, 1.0f, 0);
        k_ttt_grads<<<NC, 256, 0, stream>>>(bufA, bufB, bufC, fv, lr_f, c,
                                            hidT, a0T, a1T, a2T);
        // weight updates (fp32 accumulate); all operands pre-transposed,
        // so every fragment load is a contiguous 16B vector load.
        gemm(stream, a0T, CHUNK, bsC_, fkTc, SEQ, bsTS,
             w0f, FDIM, bsW, FDIM, FDIM, CHUNK, NFW, 1.0f, 1);
        gemm(stream, a2T, CHUNK, bsC_, fkTc, SEQ, bsTS,
             w2f, FDIM, bsW, FDIM, FDIM, CHUNK, NFW, 1.0f, 1);
        gemm(stream, a1T, CHUNK, bsC_, hidT, CHUNK, bsC_,
             w1f, FDIM, bsW, FDIM, FDIM, CHUNK, NFW, 1.0f, 1);
    }

    // ---- combine: rmsnorm(o_fast)*w + attn, cast bf16 --------------------
    k_comb<<<dim3(SEQ, NFW), 128, 0, stream>>>(o_fast, attn_o, tnw, xo);

    // ---- output projection (WMMA) ----------------------------------------
    gemm(stream, xo, HID, 0, ow_bf, HID, 0,
         out, HID, 0, SEQ, HID, HID, 1, 1.0f, 0);
}
